// CoreParamECNN_72919954751684
// MI455X (gfx1250) — compile-verified, matrix operand order checked by reading, not run on hardware
//
#include <hip/hip_runtime.h>
#include <hip/hip_bf16.h>

// ---- problem constants (from reference) ----
#define EDIM   200
#define C1N    32
#define C2N    64
#define BATCH  512
#define IHN    10
#define IWN    20
#define OWN    18        // conv1 out 8x18
#define HID    196608    // 32*64*96
#define KSPLIT 8         // K-split factor for the fused GEMM
#define NPAD   208       // 13 N-tiles of 16 covering EDIM=200

typedef __attribute__((ext_vector_type(16))) __bf16 v16bf;
typedef __attribute__((ext_vector_type(8)))  float  v8f;

union AFrag { uint4 u4[2]; v16bf v; };

// ---------------- Kernel 1: per-sample dynamic conv1 + relu ----------------
// x1[b][c1][8][18] = relu( conv(h[b] as 10x20, r1[b] as 32x3x3) )
__global__ void __launch_bounds__(256)
conv1_kernel(const float* __restrict__ h,
             const float* __restrict__ r1,
             float* __restrict__ x1) {
  __shared__ float s_img[IHN * IWN];   // 200 floats
  __shared__ float s_flt[C1N * 9];     // 288 floats
  const int b   = blockIdx.x;
  const int tid = threadIdx.x;
  if (tid < 200) s_img[tid] = h[(size_t)b * 200 + tid];
  if (tid < 288) s_flt[tid] = r1[(size_t)b * 288 + tid];
  __syncthreads();
  // 32*144 = 4608 outputs, 18 per thread
  #pragma unroll 3
  for (int k = 0; k < 18; ++k) {
    int o   = tid + 256 * k;
    int c1  = o / 144;
    int pos = o % 144;
    int oh  = pos / OWN, ow = pos % OWN;
    float acc = 0.f;
    #pragma unroll
    for (int fh = 0; fh < 3; ++fh)
      #pragma unroll
      for (int fw = 0; fw < 3; ++fw)
        acc += s_img[(oh + fh) * IWN + (ow + fw)] * s_flt[c1 * 9 + fh * 3 + fw];
    x1[(size_t)(b * C1N + c1) * 144 + pos] = fmaxf(acc, 0.f);
  }
}

// Load a 32x16 B tile of W (f32) and convert to a bf16 fragment.
__device__ __forceinline__ v16bf load_b_frag(const float* __restrict__ W,
                                             int n0, int lane, int jbase) {
  int ecol = n0 + (lane & 15);
  if (ecol > EDIM - 1) ecol = EDIM - 1;   // pad-column clamp (no OOB)
  const float* wp = W + (size_t)ecol * HID + jbase + ((lane & 16) ? 8 : 0);
  __builtin_prefetch(wp + 96, 0, 3);      // hint next c2 block of W
  float4 wv0 = *(const float4*)(wp);
  float4 wv1 = *(const float4*)(wp + 4);
  float4 wv2 = *(const float4*)(wp + 16);
  float4 wv3 = *(const float4*)(wp + 20);
  v16bf bf;
  bf[0]  = (__bf16)wv0.x; bf[1]  = (__bf16)wv0.y;
  bf[2]  = (__bf16)wv0.z; bf[3]  = (__bf16)wv0.w;
  bf[4]  = (__bf16)wv1.x; bf[5]  = (__bf16)wv1.y;
  bf[6]  = (__bf16)wv1.z; bf[7]  = (__bf16)wv1.w;
  bf[8]  = (__bf16)wv2.x; bf[9]  = (__bf16)wv2.y;
  bf[10] = (__bf16)wv2.z; bf[11] = (__bf16)wv2.w;
  bf[12] = (__bf16)wv3.x; bf[13] = (__bf16)wv3.y;
  bf[14] = (__bf16)wv3.z; bf[15] = (__bf16)wv3.w;
  return bf;
}

// -------- Kernel 2: fused conv2 (+relu) -> bf16 WMMA GEMM vs W ------------
// Output row r uses x1[r][b_lo] images and r2 rows b=(r%16)*32+b_lo.
// grid = (32 row-tiles of 16, KSPLIT);  block = 256 (8 waves of 32).
// Every wave unconditionally owns TWO N-tiles: wid and min(wid+8,12).
// Waves 5..7 redundantly recompute tile 12 (identical fragments -> identical
// results -> duplicate stores are deterministic), keeping the WMMA inner
// loop completely branch-free so EXEC stays all-ones.
__global__ void __launch_bounds__(256)
conv2_gemm_kernel(const float* __restrict__ x1,
                  const float* __restrict__ r2,
                  const float* __restrict__ W,
                  float* __restrict__ partial) {
  __shared__ float s_img[16 * 144];                 // 9216 B
  __shared__ float s_flt[16 * 576];                 // 36864 B
  __shared__ __align__(16) __bf16 s_act[16 * 96];   // 3072 B
  const int tid   = threadIdx.x;
  const int lane  = tid & 31;
  const int wid   = tid >> 5;
  const int r0    = blockIdx.x * 16;
  const int split = blockIdx.y;

  // Wave-uniform N-tile bases, forced scalar.
  const int n0a = __builtin_amdgcn_readfirstlane(wid * 16);
  const int n0b = __builtin_amdgcn_readfirstlane((wid + 8 < 13 ? wid + 8 : 12) * 16);

  v8f acc0 = {};
  v8f acc1 = {};

  for (int ii = 0; ii < 32 / KSPLIT; ++ii) {
    const int b_lo = split * (32 / KSPLIT) + ii;
    // Stage the 16 input images (x1[r0+i][b_lo], 8x18 each) and the 16 filter
    // banks (r2 row i*32+b_lo, 64 filters x 9) for this b_lo into LDS.
    for (int o = tid; o < 16 * 144; o += 256) {
      int i = o / 144, pos = o % 144;
      s_img[o] = x1[(size_t)((r0 + i) * C1N + b_lo) * 144 + pos];
    }
    for (int o = tid; o < 16 * 576; o += 256) {
      int i = o / 576, pos = o % 576;
      s_flt[o] = r2[(size_t)(i * 32 + b_lo) * 576 + pos];
    }
    __syncthreads();

    for (int c2 = 0; c2 < C2N; ++c2) {
      // conv2 for this (b_lo, c2): 16 rows x 96 spatial, 6 outputs/thread,
      // relu, pack to bf16 in LDS (A-operand staging).
      {
        const int i = tid >> 4;
        const int s_base = tid & 15;
        #pragma unroll
        for (int k = 0; k < 6; ++k) {
          int s   = s_base + 16 * k;
          int oh2 = s >> 4, ow2 = s & 15;
          float a = 0.f;
          #pragma unroll
          for (int fh = 0; fh < 3; ++fh)
            #pragma unroll
            for (int fw = 0; fw < 3; ++fw)
              a += s_img[i * 144 + (oh2 + fh) * OWN + (ow2 + fw)] *
                   s_flt[i * 576 + c2 * 9 + fh * 3 + fw];
          s_act[i * 96 + s] = (__bf16)fmaxf(a, 0.f);
        }
      }
      __syncthreads();

      #pragma unroll
      for (int t = 0; t < 3; ++t) {           // 96 = 3 x K32
        // A fragment (16x32 bf16): lane<16 holds M=lane, K{0..7,16..23};
        // lane>=16 holds M=lane-16, K{8..15,24..31}. Each octet = 16B ds load.
        AFrag af;
        const int mrow = lane & 15;
        const int koff = t * 32 + ((lane & 16) ? 8 : 0);
        af.u4[0] = *(const uint4*)&s_act[mrow * 96 + koff];
        af.u4[1] = *(const uint4*)&s_act[mrow * 96 + koff + 16];

        const int jbase = b_lo * 6144 + c2 * 96 + t * 32;
        v16bf bfa = load_b_frag(W, n0a, lane, jbase);
        acc0 = __builtin_amdgcn_wmma_f32_16x16x32_bf16(
            false, af.v, false, bfa, (short)0, acc0, false, false);
        v16bf bfb = load_b_frag(W, n0b, lane, jbase);
        acc1 = __builtin_amdgcn_wmma_f32_16x16x32_bf16(
            false, af.v, false, bfb, (short)0, acc1, false, false);
      }
      __syncthreads();   // s_act reads done before next c2 overwrites
    }
  }

  // Write partial tiles. C/D layout: VGPR v -> M = v + (lane>=16 ? 8 : 0),
  // N = lane&15. Duplicate tile-12 stores write identical data.
  {
    const int rbase = r0 + ((lane & 16) ? 8 : 0);
    const int cola = n0a + (lane & 15);
    const int colb = n0b + (lane & 15);
    float* pbase = partial + (size_t)split * BATCH * NPAD;
    #pragma unroll
    for (int v = 0; v < 8; ++v)
      pbase[(size_t)(rbase + v) * NPAD + cola] = acc0[v];
    #pragma unroll
    for (int v = 0; v < 8; ++v)
      pbase[(size_t)(rbase + v) * NPAD + colb] = acc1[v];
  }
}

// ---------------- Kernel 3: K-split reduction + bias + relu ----------------
__global__ void __launch_bounds__(256)
reduce_bias_relu(const float* __restrict__ partial,
                 const float* __restrict__ bias,
                 float* __restrict__ out) {
  int idx = blockIdx.x * 256 + threadIdx.x;
  if (idx >= BATCH * EDIM) return;
  int r = idx / EDIM, e = idx % EDIM;
  float s = bias[e];
  #pragma unroll
  for (int k = 0; k < KSPLIT; ++k)
    s += partial[((size_t)k * BATCH + r) * NPAD + e];
  out[idx] = fmaxf(s, 0.f);
}

extern "C" void kernel_launch(void* const* d_in, const int* in_sizes, int n_in,
                              void* d_out, int out_size, void* d_ws, size_t ws_size,
                              hipStream_t stream) {
  const float* h    = (const float*)d_in[0];   // (512, 200)
  const float* r1   = (const float*)d_in[1];   // (512, 288)
  const float* r2   = (const float*)d_in[2];   // (512, 576)
  const float* W    = (const float*)d_in[3];   // (200, 196608)
  const float* bias = (const float*)d_in[4];   // (200,)
  float* out = (float*)d_out;                  // (512, 200)

  // workspace layout: partial[KSPLIT][512][208] f32 (3.4 MB),
  //                   x1[512][32][144] f32 (9.4 MB)  -> ~12.3 MB total
  float* partial = (float*)d_ws;
  float* x1      = partial + (size_t)KSPLIT * BATCH * NPAD;

  conv1_kernel<<<BATCH, 256, 0, stream>>>(h, r1, x1);
  conv2_gemm_kernel<<<dim3(BATCH / 16, KSPLIT), 256, 0, stream>>>(x1, r2, W, partial);
  reduce_bias_relu<<<(BATCH * EDIM + 255) / 256, 256, 0, stream>>>(partial, bias, out);
}
